// SetTransformerEncoder_38319698215014
// MI455X (gfx1250) — matrix operand order repaired and checked by reading, not compiled, not run on hardware
//
#include <hip/hip_runtime.h>

// ---------------------------------------------------------------- constants
#define BATCH  8
#define SEQ    1024
#define MROWS  (BATCH * SEQ)   // 8192
#define DHID   512
#define NHEADS 8
#define HDIM   64              // DHID / NHEADS
#define DOUT   256
#define SM_SCALE 0.04419417382415922f   // 1/sqrt(512)

// ---------------------------------------------------------------- WMMA types
typedef __attribute__((ext_vector_type(16))) _Float16 v16h;
typedef __attribute__((ext_vector_type(8)))  float    v8f;
typedef __attribute__((ext_vector_type(4)))  unsigned u32x4;

union HalfVec {
  v16h     v;
  u32x4    q[2];
  unsigned u[8];
  _Float16 h[16];
};

__device__ __forceinline__ v8f wmma_f16(v16h a, v16h b, v8f c) {
  // D = A(16x32 f16) * B(32x16 f16) + C(16x16 f32)
  return __builtin_amdgcn_wmma_f32_16x16x32_f16(false, a, false, b, (short)0, c,
                                                false, false);
}

// ------------------------------------------------ async global -> LDS (CDNA5)
// GLOBAL_LOAD_ASYNC_TO_LDS_B128 (CDNA5 ISA op 98): per-lane 16B
// DMA into LDS, tracked with ASYNCcnt; no VGPR transit. LDS destination is the
// low 32 bits of the generic shared-pointer (LDS aperture truncation).
__device__ __forceinline__ void async_copy_b128(unsigned lds_off, const _Float16* g) {
  asm volatile("global_load_async_to_lds_b128 %0, %1, off"
               :: "v"(lds_off), "v"((unsigned long long)(size_t)g)
               : "memory");
}
__device__ __forceinline__ void async_wait0() {
  asm volatile("s_wait_asynccnt 0x0" ::: "memory");
}
__device__ __forceinline__ unsigned lds_addr(const void* p) {
  return (unsigned)(size_t)p;
}

// K-offset within a 32-wide K slice for the A operand, contiguous order to
// match ld_b16: VGPR i holds pair K = {2i, 2i+1} (+16 for lanes 16-31).
__device__ __forceinline__ int a_koff(int i, int grp) {
  return (grp << 4) | (i << 1);
}

// Gather A-operand (row = lane%16 fixed by caller through rowp, grp = lane/16)
__device__ __forceinline__ v16h ld_a16(const _Float16* rowp, int grp) {
  HalfVec u;
#pragma unroll
  for (int i = 0; i < 8; ++i)
    u.u[i] = *(const unsigned*)(rowp + a_koff(i, grp));
  return u.v;
}

// Gather B-operand: 16 contiguous halves (K = h + 16*(lane/16), N = lane%16),
// caller passes p already offset by lane-dependent row/column.
__device__ __forceinline__ v16h ld_b16(const _Float16* p) {
  HalfVec u;
  u.q[0] = ((const u32x4*)p)[0];
  u.q[1] = ((const u32x4*)p)[1];
  return u.v;
}

// ---------------------------------------------------------------- f32 -> f16
__global__ void k_cvt(const float* __restrict__ s, _Float16* __restrict__ d, int n) {
  const int i = blockIdx.x * blockDim.x + threadIdx.x;
  if (i < n) d[i] = (_Float16)s[i];
}

// ---------------------------------------------------------------- WMMA GEMM
// C[M,512] = f(A[M,K] @ W[K,512] + bias), A/W f16, accum f32.
// mode 0: out = acc + bias              (projection)
// mode 1: out = resid + relu(acc+bias)  (MAB output projection)
// Block tile 128(M) x 64(N), 4 waves; K staged through double-buffered LDS in
// chunks of 32: the A tile arrives via async-to-LDS DMA (overlapped with the
// scalar transposed W staging and with the previous chunk's WMMAs).
__global__ __launch_bounds__(128) void k_gemm(
    const _Float16* __restrict__ A, int lda,
    const _Float16* __restrict__ W,
    const float* __restrict__ bias,
    const float* __restrict__ resid,
    float* __restrict__ outf32,
    _Float16* __restrict__ outf16,
    int M, int K, int mode)
{
  __shared__ alignas(16) _Float16 Alds[2][128][32];   // 16 KB
  __shared__ alignas(16) _Float16 Wlds[2][64][32];    //  8 KB, transposed [n][k]

  const int tid  = threadIdx.x;
  const int wave = tid >> 5;
  const int lane = tid & 31;
  const int grp  = lane >> 4;
  const int l16  = lane & 15;
  const int m0 = blockIdx.x * 128;
  const int n0 = blockIdx.y * 64;

  v8f acc[2][4];
#pragma unroll
  for (int a = 0; a < 2; ++a)
#pragma unroll
    for (int c = 0; c < 4; ++c)
#pragma unroll
      for (int r = 0; r < 8; ++r) acc[a][c][r] = 0.0f;

  auto stage = [&](int k0, int buf) {
    // A tile: one 32-half row per thread, 4 x async b128 DMA straight to LDS
    const _Float16* src = A + (size_t)(m0 + tid) * lda + k0;
    const unsigned  dst = lds_addr(&Alds[buf][tid][0]);
    async_copy_b128(dst,      src);
    async_copy_b128(dst + 16, src + 8);
    async_copy_b128(dst + 32, src + 16);
    async_copy_b128(dst + 48, src + 24);
    // W tile transposed (scalar ds stores; overlaps the async DMA above)
    const int n  = tid & 63;
    const int kh = (tid >> 6) << 4;   // 0 or 16
    const _Float16* ws = W + (size_t)(k0 + kh) * DHID + (n0 + n);
#pragma unroll
    for (int j = 0; j < 16; ++j)
      Wlds[buf][n][kh + j] = ws[(size_t)j * DHID];
  };

  stage(0, 0);
  async_wait0();
  __syncthreads();

  const int nk = K >> 5;
  for (int t = 0; t < nk; ++t) {
    const int buf = t & 1;
    if (t + 1 < nk) stage((t + 1) << 5, buf ^ 1);   // prefetch next chunk

    const v16h a0 = ld_a16(&Alds[buf][wave * 32 + l16][0], grp);
    const v16h a1 = ld_a16(&Alds[buf][wave * 32 + 16 + l16][0], grp);
#pragma unroll
    for (int ns = 0; ns < 4; ++ns) {
      const v16h b = ld_b16(&Wlds[buf][ns * 16 + l16][grp * 16]);
      acc[0][ns] = wmma_f16(a0, b, acc[0][ns]);
      acc[1][ns] = wmma_f16(a1, b, acc[1][ns]);
    }
    async_wait0();     // own-wave ASYNCcnt drained before the barrier
    __syncthreads();   // (barrier itself covers the ds stores via dscnt wait)
  }

  // epilogue: C layout is row = r + 8*grp, col = lane%16 per 16x16 tile
#pragma unroll
  for (int ms = 0; ms < 2; ++ms)
#pragma unroll
    for (int ns = 0; ns < 4; ++ns)
#pragma unroll
      for (int r = 0; r < 8; ++r) {
        const int row = m0 + wave * 32 + ms * 16 + r + grp * 8;
        const int col = n0 + ns * 16 + l16;
        const size_t off = (size_t)row * DHID + col;
        float v = acc[ms][ns][r] + bias[col];
        if (mode == 1) v = resid[off] + fmaxf(v, 0.0f);
        if (outf32) outf32[off] = v;
        outf16[off] = (_Float16)v;
      }
}

// ---------------------------------------------------------------- attention
// Flash attention, one block per (batch, head, 64 q rows); 4 waves, each wave
// owns 16 q rows. K/V streamed in double-buffered 32-row tiles: K arrives via
// async-to-LDS DMA, V is transposed into LDS with scalar stores (overlapped).
// Softmax uses 16-lane butterfly reductions matching the C-matrix layout; P is
// re-laid-out C->A through per-wave LDS scratch (same-wave LDS ops in-order).
// O = q + softmax(qK^T/sqrt(512)) V.
__global__ __launch_bounds__(128) void k_attn(
    const _Float16* __restrict__ q16,
    const _Float16* __restrict__ k16,
    const _Float16* __restrict__ v16,
    const float* __restrict__ q32,
    float* __restrict__ O32,
    _Float16* __restrict__ O16)
{
  __shared__ alignas(16) _Float16 Klds[2][32][64];   // [k_local][d]     8 KB
  __shared__ alignas(16) _Float16 Vt[2][64][32];     // [d][k_local]     8 KB
  __shared__ alignas(16) _Float16 Plds[4][16][32];   // per-wave P       4 KB

  const int tid  = threadIdx.x;
  const int wave = tid >> 5;
  const int lane = tid & 31;
  const int grp  = lane >> 4;
  const int l16  = lane & 15;
  const int b = blockIdx.z, h = blockIdx.y;
  const int qb = blockIdx.x * 64 + wave * 16;
  const size_t base = ((size_t)b * SEQ) * DHID + (size_t)h * HDIM;

  v16h aq0, aq1;
  {
    const _Float16* rowp = q16 + base + (size_t)(qb + l16) * DHID;
    aq0 = ld_a16(rowp, grp);        // d = 0..31
    aq1 = ld_a16(rowp + 32, grp);   // d = 32..63
  }

  float m[8], l[8], alpha[8];
  v8f acc[4];
#pragma unroll
  for (int r = 0; r < 8; ++r) { m[r] = -1e30f; l[r] = 0.0f; }
#pragma unroll
  for (int dt = 0; dt < 4; ++dt)
#pragma unroll
    for (int r = 0; r < 8; ++r) acc[dt][r] = 0.0f;

  auto stage = [&](int kt, int buf) {
    const int r  = tid >> 2;
    const int ds = (tid & 3) << 4;
    // K tile row-major: async b128 DMA straight into LDS
    const _Float16* ks = k16 + base + (size_t)(kt + r) * DHID + ds;
    const unsigned  kd = lds_addr(&Klds[buf][r][ds]);
    async_copy_b128(kd,      ks);
    async_copy_b128(kd + 16, ks + 8);
    // V tile transposed with scalar stores (overlaps the async DMA)
    const _Float16* vs = v16 + base + (size_t)(kt + r) * DHID + ds;
#pragma unroll
    for (int j = 0; j < 16; ++j) Vt[buf][ds + j][r] = vs[j];
  };

  stage(0, 0);
  async_wait0();
  __syncthreads();

  for (int t = 0; t < SEQ / 32; ++t) {
    const int buf = t & 1;
    if (t + 1 < SEQ / 32) stage((t + 1) * 32, buf ^ 1);   // prefetch next tile

    // scores: two 16-wide k sub-tiles, K split d = 0..31 / 32..63
    v8f c0, c1;
#pragma unroll
    for (int r = 0; r < 8; ++r) { c0[r] = 0.0f; c1[r] = 0.0f; }
    c0 = wmma_f16(aq0, ld_b16(&Klds[buf][l16][grp * 16]), c0);
    c0 = wmma_f16(aq1, ld_b16(&Klds[buf][l16][32 + grp * 16]), c0);
    c1 = wmma_f16(aq0, ld_b16(&Klds[buf][16 + l16][grp * 16]), c1);
    c1 = wmma_f16(aq1, ld_b16(&Klds[buf][16 + l16][32 + grp * 16]), c1);

    // streaming softmax per q row (row r lives in lanes {grp*16..grp*16+15})
#pragma unroll
    for (int r = 0; r < 8; ++r) {
      const float v0 = c0[r] * SM_SCALE, v1 = c1[r] * SM_SCALE;
      float tmx = fmaxf(v0, v1);
#pragma unroll
      for (int off = 8; off > 0; off >>= 1) tmx = fmaxf(tmx, __shfl_xor(tmx, off, 32));
      const float nm = fmaxf(m[r], tmx);
      const float a  = __expf(m[r] - nm);
      const float p0 = __expf(v0 - nm);
      const float p1 = __expf(v1 - nm);
      float rs = p0 + p1;
#pragma unroll
      for (int off = 8; off > 0; off >>= 1) rs += __shfl_xor(rs, off, 32);
      l[r] = l[r] * a + rs;
      m[r] = nm;
      alpha[r] = a;
      Plds[wave][r + grp * 8][l16]      = (_Float16)p0;
      Plds[wave][r + grp * 8][16 + l16] = (_Float16)p1;
    }
#pragma unroll
    for (int dt = 0; dt < 4; ++dt)
#pragma unroll
      for (int r = 0; r < 8; ++r) acc[dt][r] *= alpha[r];

    // P (16x32) back in A layout; same-wave LDS ops are in-order, no barrier
    const v16h ap = ld_a16(&Plds[wave][l16][0], grp);
#pragma unroll
    for (int dt = 0; dt < 4; ++dt)
      acc[dt] = wmma_f16(ap, ld_b16(&Vt[buf][dt * 16 + l16][grp * 16]), acc[dt]);

    async_wait0();     // next tile's DMA drained (own wave) before the barrier
    __syncthreads();   // protects both LDS buffers across waves
  }

#pragma unroll
  for (int dt = 0; dt < 4; ++dt)
#pragma unroll
    for (int r = 0; r < 8; ++r) {
      const int row = qb + r + grp * 8;
      const int col = h * HDIM + dt * 16 + l16;
      const size_t off = ((size_t)b * SEQ + row) * DHID + col;
      const float v = q32[off] + acc[dt][r] / l[r];   // residual on head-split q
      O32[off] = v;
      O16[off] = (_Float16)v;
    }
}

// ---------------------------------------------------------------- PMA (Sq=1)
__global__ __launch_bounds__(256) void k_pma_qproj(
    const float* __restrict__ S, const float* __restrict__ Wq,
    const float* __restrict__ bq, float* __restrict__ qp)
{
  const int j = blockIdx.x * 256 + threadIdx.x;
  if (j < DHID) {
    float a = bq[j];
    for (int i = 0; i < DHID; ++i) a += S[i] * Wq[(size_t)i * DHID + j];
    qp[j] = a;
  }
}

__global__ __launch_bounds__(256) void k_pma_attn(
    const float* __restrict__ qp, const _Float16* __restrict__ k16,
    const _Float16* __restrict__ v16, float* __restrict__ P)
{
  __shared__ float sc[SEQ];
  __shared__ float red[256];
  __shared__ float part[4][64];
  const int h = blockIdx.x, b = blockIdx.y, tid = threadIdx.x;
  const size_t base = ((size_t)b * SEQ) * DHID + (size_t)h * HDIM;

  float lm = -1e30f;
  for (int s = tid; s < SEQ; s += 256) {
    const _Float16* kr = k16 + base + (size_t)s * DHID;
    float a = 0.0f;
    for (int d = 0; d < HDIM; ++d) a += qp[h * HDIM + d] * (float)kr[d];
    a *= SM_SCALE;
    sc[s] = a;
    lm = fmaxf(lm, a);
  }
  red[tid] = lm; __syncthreads();
  for (int st = 128; st > 0; st >>= 1) {
    if (tid < st) red[tid] = fmaxf(red[tid], red[tid + st]);
    __syncthreads();
  }
  const float mx = red[0];
  __syncthreads();
  float ls = 0.0f;
  for (int s = tid; s < SEQ; s += 256) {
    const float pv = __expf(sc[s] - mx);
    sc[s] = pv;
    ls += pv;
  }
  red[tid] = ls; __syncthreads();
  for (int st = 128; st > 0; st >>= 1) {
    if (tid < st) red[tid] += red[tid + st];
    __syncthreads();
  }
  const float inv = 1.0f / red[0];

  const int d = tid & 63, ch = tid >> 6;
  float a = 0.0f;
  for (int s = ch * 256; s < ch * 256 + 256; ++s)
    a += sc[s] * (float)v16[base + (size_t)s * DHID + d];
  part[ch][d] = a; __syncthreads();
  if (tid < 64)
    P[(size_t)b * DHID + h * HDIM + tid] =
        qp[h * HDIM + tid] +
        (part[0][tid] + part[1][tid] + part[2][tid] + part[3][tid]) * inv;
}

__global__ __launch_bounds__(512) void k_pma_out(
    const float* __restrict__ P, const float* __restrict__ Wo,
    const float* __restrict__ bo, float* __restrict__ Hp)
{
  const int b = blockIdx.x, j = threadIdx.x;
  const float* pr = P + (size_t)b * DHID;
  float a = bo[j];
  for (int i = 0; i < DHID; ++i) a += pr[i] * Wo[(size_t)i * DHID + j];
  Hp[(size_t)b * DHID + j] = pr[j] + fmaxf(a, 0.0f);
}

__global__ __launch_bounds__(256) void k_final(
    const float* __restrict__ Hp, const float* __restrict__ Wf,
    const float* __restrict__ bf, float* __restrict__ out)
{
  const int b = blockIdx.x, j = threadIdx.x;
  const float* pr = Hp + (size_t)b * DHID;
  float a = bf[j];
  for (int i = 0; i < DHID; ++i) a += pr[i] * Wf[(size_t)i * DOUT + j];
  out[(size_t)b * DOUT + j] = fmaxf(a, 0.0f);
}

// ---------------------------------------------------------------- host side
extern "C" void kernel_launch(void* const* d_in, const int* in_sizes, int n_in,
                              void* d_out, int out_size, void* d_ws, size_t ws_size,
                              hipStream_t stream)
{
  (void)out_size; (void)ws_size;
  struct Mab { const float *Wq,*bq,*Wk,*bk,*Wv,*bv,*Wo,*bo; };
  const float* X = (const float*)d_in[0];
  const float *pmaS = nullptr, *Wf = nullptr, *bf = nullptr;
  Mab sab[4], pma;

  // Detect flattening order: JAX pytree (sorted keys) => in_sizes[2] == |bf| = 256,
  // Python insertion order => in_sizes[2] == |sabs[0].bq| = 512.
  const bool sorted = (n_in > 2 && in_sizes[2] == 256);
  if (sorted) {
    Wf = (const float*)d_in[1]; bf = (const float*)d_in[2];
    pmaS = (const float*)d_in[3];
    auto fill = [&](Mab& m, int o) {
      m.Wk = (const float*)d_in[o + 0]; m.Wo = (const float*)d_in[o + 1];
      m.Wq = (const float*)d_in[o + 2]; m.Wv = (const float*)d_in[o + 3];
      m.bk = (const float*)d_in[o + 4]; m.bo = (const float*)d_in[o + 5];
      m.bq = (const float*)d_in[o + 6]; m.bv = (const float*)d_in[o + 7];
    };
    fill(pma, 4);
    for (int i = 0; i < 4; ++i) fill(sab[i], 12 + 8 * i);
  } else {
    auto fill = [&](Mab& m, int o) {
      m.Wq = (const float*)d_in[o + 0]; m.bq = (const float*)d_in[o + 1];
      m.Wk = (const float*)d_in[o + 2]; m.bk = (const float*)d_in[o + 3];
      m.Wv = (const float*)d_in[o + 4]; m.bv = (const float*)d_in[o + 5];
      m.Wo = (const float*)d_in[o + 6]; m.bo = (const float*)d_in[o + 7];
    };
    for (int i = 0; i < 4; ++i) fill(sab[i], 1 + 8 * i);
    pmaS = (const float*)d_in[33];
    fill(pma, 34);
    Wf = (const float*)d_in[42]; bf = (const float*)d_in[43];
  }

  // -------- workspace carve (~85 MB total)
  char* wsp = (char*)d_ws; size_t off = 0;
  auto carve = [&](size_t bytes) -> void* {
    void* r = wsp + off; off += (bytes + 255) & ~(size_t)255; return r;
  };
  const size_t MD = (size_t)MROWS * DHID;
  _Float16* Hf16 = (_Float16*)carve(MD * 2);
  float*    qf32 = (float*)   carve(MD * 4);
  _Float16* qf16 = (_Float16*)carve(MD * 2);
  _Float16* kf16 = (_Float16*)carve(MD * 2);
  _Float16* vf16 = (_Float16*)carve(MD * 2);
  float*    Of32 = (float*)   carve(MD * 4);
  _Float16* Of16 = (_Float16*)carve(MD * 2);
  const int sabK[4] = {256, 512, 512, 512};
  _Float16 *wq16[4], *wk16[4], *wv16[4], *wo16[4];
  for (int i = 0; i < 4; ++i) {
    wq16[i] = (_Float16*)carve((size_t)sabK[i] * DHID * 2);
    wk16[i] = (_Float16*)carve((size_t)sabK[i] * DHID * 2);
    wv16[i] = (_Float16*)carve((size_t)sabK[i] * DHID * 2);
    wo16[i] = (_Float16*)carve((size_t)DHID * DHID * 2);
  }
  _Float16* pwk16 = (_Float16*)carve((size_t)DHID * DHID * 2);
  _Float16* pwv16 = (_Float16*)carve((size_t)DHID * DHID * 2);
  float* qp = (float*)carve(DHID * 4);
  float* P  = (float*)carve((size_t)BATCH * DHID * 4);
  float* Hp = (float*)carve((size_t)BATCH * DHID * 4);

  auto cvt = [&](const float* s, _Float16* d, int n) {
    k_cvt<<<dim3((n + 255) / 256), dim3(256), 0, stream>>>(s, d, n);
  };
  auto gemm = [&](const _Float16* A, int lda, const _Float16* W, const float* bias,
                  const float* resid, float* o32, _Float16* o16, int K, int mode) {
    k_gemm<<<dim3(MROWS / 128, DHID / 64), dim3(128), 0, stream>>>(
        A, lda, W, bias, resid, o32, o16, MROWS, K, mode);
  };

  // weight + input conversion to f16
  cvt(X, Hf16, MROWS * 256);
  for (int i = 0; i < 4; ++i) {
    cvt(sab[i].Wq, wq16[i], sabK[i] * DHID);
    cvt(sab[i].Wk, wk16[i], sabK[i] * DHID);
    cvt(sab[i].Wv, wv16[i], sabK[i] * DHID);
    cvt(sab[i].Wo, wo16[i], DHID * DHID);
  }
  cvt(pma.Wk, pwk16, DHID * DHID);
  cvt(pma.Wv, pwv16, DHID * DHID);

  // 4 x SAB
  for (int i = 0; i < 4; ++i) {
    const int K = sabK[i];
    gemm(Hf16, K, wq16[i], sab[i].bq, nullptr, qf32, qf16, K, 0);
    gemm(Hf16, K, wk16[i], sab[i].bk, nullptr, nullptr, kf16, K, 0);
    gemm(Hf16, K, wv16[i], sab[i].bv, nullptr, nullptr, vf16, K, 0);
    k_attn<<<dim3(SEQ / 64, NHEADS, BATCH), dim3(128), 0, stream>>>(
        qf16, kf16, vf16, qf32, Of32, Of16);
    gemm(Of16, DHID, wo16[i], sab[i].bo, Of32, nullptr, Hf16, DHID, 1);
  }

  // PMA (num_seeds = 1) + final linear
  gemm(Hf16, DHID, pwk16, pma.bk, nullptr, nullptr, kf16, DHID, 0);
  gemm(Hf16, DHID, pwv16, pma.bv, nullptr, nullptr, vf16, DHID, 0);
  k_pma_qproj<<<dim3(2), dim3(256), 0, stream>>>(pmaS, pma.Wq, pma.bq, qp);
  k_pma_attn<<<dim3(NHEADS, BATCH), dim3(256), 0, stream>>>(qp, kf16, vf16, P);
  k_pma_out<<<dim3(BATCH), dim3(512), 0, stream>>>(P, pma.Wo, pma.bo, Hp);
  k_final<<<dim3(BATCH), dim3(256), 0, stream>>>(Hp, Wf, bf, (float*)d_out);
}
